// CrossAttentionBlock_74526272520385
// MI455X (gfx1250) — matrix operand order repaired and checked by reading, not compiled, not run on hardware
//
#include <hip/hip_runtime.h>

// ---------------------------------------------------------------------------
// Types / helpers
// ---------------------------------------------------------------------------
typedef __attribute__((ext_vector_type(16))) __bf16 bf16x16;
typedef __attribute__((ext_vector_type(8)))  float  f32x8;

union FragB16 { uint4 u[2]; bf16x16 v; };

__device__ __forceinline__ unsigned short f2bf(float f) {
  unsigned int u = __float_as_uint(f);
  unsigned int r = (u + 0x7FFFu + ((u >> 16) & 1u)) >> 16;
  return (unsigned short)r;
}

__device__ __forceinline__ f32x8 zero8() {
  f32x8 z = {0.f, 0.f, 0.f, 0.f, 0.f, 0.f, 0.f, 0.f};
  return z;
}

__device__ __forceinline__ f32x8 wmma_bf16(bf16x16 a, bf16x16 b, f32x8 c) {
  // (neg_a, A, neg_b, B, c_mod, C, reuse_a, reuse_b)
  return __builtin_amdgcn_wmma_f32_16x16x32_bf16(false, a, false, b, (short)0, c,
                                                 false, false);
}

// max-reduce across the 16 lanes of each wave half via ds_swizzle immediates
// (group-of-32 mode: imm = xor_mask<<10 | and_mask 0x1f)
__device__ __forceinline__ float redmax16(float x) {
  x = fmaxf(x, __int_as_float(__builtin_amdgcn_ds_swizzle(__float_as_int(x), 0x201F)));
  x = fmaxf(x, __int_as_float(__builtin_amdgcn_ds_swizzle(__float_as_int(x), 0x101F)));
  x = fmaxf(x, __int_as_float(__builtin_amdgcn_ds_swizzle(__float_as_int(x), 0x081F)));
  x = fmaxf(x, __int_as_float(__builtin_amdgcn_ds_swizzle(__float_as_int(x), 0x041F)));
  return x;
}

// Async 32-byte copy (two B128 lanes-worth) global -> LDS, tracked by ASYNCcnt.
// the immediate offset is added to both the LDS address and the global address
__device__ __forceinline__ void async_copy32(unsigned lds, unsigned long long ga) {
  asm volatile("global_load_async_to_lds_b128 %0, %1, off\n\t"
               "global_load_async_to_lds_b128 %0, %1, off offset:16"
               :: "v"(lds), "v"(ga) : "memory");
}
__device__ __forceinline__ void wait_async_le4() {
  asm volatile("s_wait_asynccnt 0x4" ::: "memory");
}
__device__ __forceinline__ void wait_async_0() {
  asm volatile("s_wait_asynccnt 0x0" ::: "memory");
}

// ---------------------------------------------------------------------------
// Weight transpose + f32 -> bf16 :  wt[n*K + k] = bf16(w[k*N + n])
// ---------------------------------------------------------------------------
__global__ __launch_bounds__(256) void convT_kernel(const float* __restrict__ w,
                                                    unsigned short* __restrict__ wt,
                                                    int K, int N) {
  size_t idx = (size_t)blockIdx.x * 256 + threadIdx.x;
  size_t total = (size_t)K * N;
  if (idx >= total) return;
  int k = (int)(idx % K);
  int n = (int)(idx / K);
  wt[idx] = f2bf(w[(size_t)k * N + n]);
}

// ---------------------------------------------------------------------------
// LayerNorm (D = 512) with optional per-batch broadcast add, bf16 output
// ---------------------------------------------------------------------------
__global__ __launch_bounds__(256) void ln_kernel(const float* __restrict__ in,
                                                 const float* __restrict__ g,
                                                 const float* __restrict__ be,
                                                 const float* __restrict__ add,
                                                 int rows_per_batch,
                                                 unsigned short* __restrict__ out) {
  const int D = 512;
  int row = blockIdx.x, t = threadIdx.x;
  __shared__ float red[256];
  const float* rp = in + (size_t)row * D;
  float x0 = rp[t], x1 = rp[t + 256];
  if (add) {
    int bb = row / rows_per_batch;
    x0 += add[bb * D + t];
    x1 += add[bb * D + t + 256];
  }
  red[t] = x0 + x1;
  __syncthreads();
  for (int o = 128; o > 0; o >>= 1) { if (t < o) red[t] += red[t + o]; __syncthreads(); }
  float mu = red[0] * (1.f / 512.f);
  __syncthreads();
  float d0 = x0 - mu, d1 = x1 - mu;
  red[t] = d0 * d0 + d1 * d1;
  __syncthreads();
  for (int o = 128; o > 0; o >>= 1) { if (t < o) red[t] += red[t + o]; __syncthreads(); }
  float rstd = rsqrtf(red[0] * (1.f / 512.f) + 1e-5f);
  out[(size_t)row * D + t]       = f2bf(d0 * rstd * g[t] + be[t]);
  out[(size_t)row * D + t + 256] = f2bf(d1 * rstd * g[t + 256] + be[t + 256]);
}

// ---------------------------------------------------------------------------
// Time embedding MLP: ti[b,512] = swish(t@tw1+tb1)@tw2 + tb2   (B=8, tiny)
// ---------------------------------------------------------------------------
__global__ __launch_bounds__(256) void time_kernel(const float* __restrict__ t,
                                                   const float* __restrict__ tw1,
                                                   const float* __restrict__ tb1,
                                                   const float* __restrict__ tw2,
                                                   const float* __restrict__ tb2,
                                                   float* __restrict__ ti) {
  __shared__ float th[256];
  int b = blockIdx.x, j = threadIdx.x;
  float s = tb1[j];
  for (int k = 0; k < 256; k++) s += t[b * 256 + k] * tw1[k * 256 + j];
  th[j] = s / (1.f + __expf(-s));
  __syncthreads();
  for (int i = j; i < 512; i += 256) {
    float s2 = tb2[i];
    for (int k = 0; k < 256; k++) s2 += th[k] * tw2[k * 512 + i];
    ti[b * 512 + i] = s2;
  }
}

// ---------------------------------------------------------------------------
// V transpose: vt[((b*8+h)*512 + c)*1024 + m] = kv[(b*1024+m)*4608 + 512 + h*512 + c]
// ---------------------------------------------------------------------------
__global__ __launch_bounds__(256) void vtrans_kernel(const unsigned short* __restrict__ kv,
                                                     unsigned short* __restrict__ vt) {
  size_t idx = (size_t)blockIdx.x * 256 + threadIdx.x;  // < 8*8*512*1024
  int m = (int)(idx & 1023);
  int c = (int)((idx >> 10) & 511);
  int h = (int)((idx >> 19) & 7);
  int b = (int)(idx >> 22);
  vt[idx] = kv[(size_t)(b * 1024 + m) * 4608 + 512 + h * 512 + c];
}

// ---------------------------------------------------------------------------
// Generic bf16 WMMA GEMM: C[M,N] = A[M,K](bf16) * Bt[N,K](bf16)^T + bias
//   ACT: 0=none 1=swish ; OUTBF: bf16/f32 output ; HASRES: += res[M,N] (f32)
//   128x128 block tile, K-step 32, 8 waves, each wave 32x64 (2x4 WMMA frags).
//   Double-buffered LDS staging via GLOBAL_LOAD_ASYNC_TO_LDS_B128 (ASYNCcnt).
// ---------------------------------------------------------------------------
template <int ACT, int OUTBF, int HASRES>
__global__ __launch_bounds__(256) void gemm_bf16_kernel(
    const unsigned short* __restrict__ A, const unsigned short* __restrict__ Bt,
    const float* __restrict__ bias, const float* __restrict__ res,
    void* __restrict__ outp, int M, int N, int K) {
  __shared__ __align__(16) unsigned short As[2][128 * 40];
  __shared__ __align__(16) unsigned short Bs[2][128 * 40];
  const int tid = threadIdx.x;
  const int lane = tid & 31, wave = tid >> 5;
  const int half = lane >> 4, lm = lane & 15;
  const int wr = wave >> 1, wc = wave & 1;
  const int m0 = blockIdx.y * 128, n0 = blockIdx.x * 128;

  const int srow = tid >> 1, spart = (tid & 1) * 16;
  const unsigned short* gA = A + (size_t)(m0 + srow) * K + spart;
  const unsigned short* gB = Bt + (size_t)(n0 + srow) * K + spart;
  unsigned ldsA[2], ldsB[2];
  ldsA[0] = (unsigned)(uintptr_t)&As[0][srow * 40 + spart];
  ldsA[1] = (unsigned)(uintptr_t)&As[1][srow * 40 + spart];
  ldsB[0] = (unsigned)(uintptr_t)&Bs[0][srow * 40 + spart];
  ldsB[1] = (unsigned)(uintptr_t)&Bs[1][srow * 40 + spart];

  f32x8 acc[2][4];
#pragma unroll
  for (int i = 0; i < 2; i++)
#pragma unroll
    for (int j = 0; j < 4; j++) acc[i][j] = zero8();

  const int nk = K >> 5;
  // prologue: stage 0 into buffer 0 (4 async B128 per thread per stage)
  async_copy32(ldsA[0], (unsigned long long)(uintptr_t)gA);
  async_copy32(ldsB[0], (unsigned long long)(uintptr_t)gB);

  for (int kt = 0; kt < nk; kt++) {
    const int cur = kt & 1;
    if (kt + 1 < nk) {
      // previous trailing barrier guarantees buffer cur^1 is no longer read
      async_copy32(ldsA[cur ^ 1], (unsigned long long)(uintptr_t)(gA + (size_t)(kt + 1) * 32));
      async_copy32(ldsB[cur ^ 1], (unsigned long long)(uintptr_t)(gB + (size_t)(kt + 1) * 32));
      wait_async_le4();  // async completes in order: <=4 outstanding => stage kt landed
    } else {
      wait_async_0();
    }
    __syncthreads();

    FragB16 af[2], bfr[4];
#pragma unroll
    for (int i = 0; i < 2; i++) {
      const unsigned short* p = &As[cur][(wr * 32 + i * 16 + lm) * 40 + half * 8];
      af[i].u[0] = *(const uint4*)p;
      af[i].u[1] = *(const uint4*)(p + 16);
    }
#pragma unroll
    for (int j = 0; j < 4; j++) {
      const unsigned short* p = &Bs[cur][(wc * 64 + j * 16 + lm) * 40 + half * 8];
      bfr[j].u[0] = *(const uint4*)p;
      bfr[j].u[1] = *(const uint4*)(p + 16);
    }
#pragma unroll
    for (int i = 0; i < 2; i++)
#pragma unroll
      for (int j = 0; j < 4; j++)
        acc[i][j] = wmma_bf16(af[i].v, bfr[j].v, acc[i][j]);
    __syncthreads();
  }

#pragma unroll
  for (int i = 0; i < 2; i++) {
#pragma unroll
    for (int j = 0; j < 4; j++) {
      int col = n0 + wc * 64 + j * 16 + lm;
      float bv = bias ? bias[col] : 0.f;
#pragma unroll
      for (int r = 0; r < 8; r++) {
        int row = m0 + wr * 32 + i * 16 + r + 8 * half;
        float v = acc[i][j][r] + bv;
        if (HASRES) v += res[(size_t)row * N + col];
        if (ACT == 1) v = v / (1.f + __expf(-v));  // swish
        if (OUTBF)
          ((unsigned short*)outp)[(size_t)row * N + col] = f2bf(v);
        else
          ((float*)outp)[(size_t)row * N + col] = v;
      }
    }
  }
}

// ---------------------------------------------------------------------------
// Flash cross-attention. Block = 128 thr (4 waves) per (b, h, 16-row q tile).
// Each wave: redundant 16x32 score chunk + online softmax, then P*V over its
// private 128-channel slice. Row sums come from an extra WMMA against an
// all-ones B fragment (P @ 1) so only the max needs a lane reduction.
//   q  : [B,1024,512] bf16 (cols = h*64 + dh)
//   kv : [B,1024,4608] bf16 (k part = cols h*64+dh)
//   vt : [B,H,512,1024] bf16 (channel-major, m contiguous)
//   out: [B,1024,4096] bf16 (cols = h*512 + c)
// ---------------------------------------------------------------------------
__global__ __launch_bounds__(128) void attn_kernel(
    const unsigned short* __restrict__ q, const unsigned short* __restrict__ kv,
    const unsigned short* __restrict__ vt, unsigned short* __restrict__ out) {
  constexpr int Bn = 1024, Mm = 1024;
  __shared__ __align__(16) unsigned short Plds[4][512];
  int bid = blockIdx.x;
  int nt = bid & 63;
  int h = (bid >> 6) & 7;
  int b = bid >> 9;
  int tid = threadIdx.x;
  int wave = tid >> 5, lane = tid & 31;
  int half = lane >> 4, lm = lane & 15;
  int n0 = nt * 16;
  int c0 = wave * 128;

  // q A-fragments (16 rows x 32 dh) for dh chunks 0..31 and 32..63
  FragB16 qf[2];
#pragma unroll
  for (int s = 0; s < 2; s++) {
    const unsigned short* p =
        q + (size_t)(b * Bn + n0 + lm) * 512 + h * 64 + s * 32 + half * 8;
    qf[s].u[0] = *(const uint4*)p;
    qf[s].u[1] = *(const uint4*)(p + 16);
  }

  // all-ones bf16 B fragment (for P @ 1 row sums)
  FragB16 ones;
#pragma unroll
  for (int i = 0; i < 2; i++) {
    ones.u[i].x = 0x3F803F80u; ones.u[i].y = 0x3F803F80u;
    ones.u[i].z = 0x3F803F80u; ones.u[i].w = 0x3F803F80u;
  }

  float mrun[8];
  f32x8 acc[8];
  f32x8 sumacc = zero8();
#pragma unroll
  for (int r = 0; r < 8; r++) mrun[r] = -1e30f;
#pragma unroll
  for (int f = 0; f < 8; f++) acc[f] = zero8();

  unsigned short* P = &Plds[wave][0];

  for (int mc = 0; mc < Mm; mc += 32) {
    // --- scores S[16 x 32] = q @ k^T, two 16-col tiles, K=64 (2 WMMA each)
    f32x8 S[2];
#pragma unroll
    for (int s = 0; s < 2; s++) {
      f32x8 c = zero8();
#pragma unroll
      for (int ks = 0; ks < 2; ks++) {
        FragB16 kf;
        const unsigned short* p = kv +
            (size_t)(b * Mm + mc + s * 16 + lm) * 4608 + h * 64 + ks * 32 + half * 8;
        kf.u[0] = *(const uint4*)p;
        kf.u[1] = *(const uint4*)(p + 16);
        c = wmma_bf16(qf[ks].v, kf.v, c);
      }
      S[s] = c;
    }

    // --- online softmax (max only; sums accumulated by WMMA below)
    float scl[8];
#pragma unroll
    for (int r = 0; r < 8; r++) {
      float s0 = S[0][r] * 0.125f, s1 = S[1][r] * 0.125f;  // * DH^-0.5
      float mx = redmax16(fmaxf(s0, s1));
      float mnew = fmaxf(mrun[r], mx);
      float p0 = __expf(s0 - mnew), p1 = __expf(s1 - mnew);
      scl[r] = __expf(mrun[r] - mnew);
      mrun[r] = mnew;
      // C-layout -> row-major P patch in per-wave LDS (16 x 32)
      P[(r + 8 * half) * 32 + lm] = f2bf(p0);
      P[(r + 8 * half) * 32 + 16 + lm] = f2bf(p1);
    }
#pragma unroll
    for (int f = 0; f < 8; f++)
#pragma unroll
      for (int r = 0; r < 8; r++) acc[f][r] *= scl[r];
#pragma unroll
    for (int r = 0; r < 8; r++) sumacc[r] *= scl[r];

    // reload P as an A-fragment (16 x 32)
    FragB16 pf;
    {
      const unsigned short* pp = P + lm * 32 + half * 8;
      pf.u[0] = *(const uint4*)pp;
      pf.u[1] = *(const uint4*)(pp + 16);
    }

    // row sums: sumacc += P @ ones  (every output column = sum_m P[n,m])
    sumacc = wmma_bf16(pf.v, ones.v, sumacc);

    // --- acc[16 x 128] += P[16 x 32] @ V[32 x 128] (8 WMMA)
#pragma unroll
    for (int f = 0; f < 8; f++) {
      FragB16 vf;
      const unsigned short* p = vt +
          ((size_t)((b * 8 + h) * 512 + c0 + f * 16 + lm)) * Mm + mc + half * 8;
      vf.u[0] = *(const uint4*)p;
      vf.u[1] = *(const uint4*)(p + 16);
      acc[f] = wmma_bf16(pf.v, vf.v, acc[f]);
    }
  }

  // --- normalize and store
#pragma unroll
  for (int f = 0; f < 8; f++) {
#pragma unroll
    for (int r = 0; r < 8; r++) {
      int row = n0 + r + 8 * half;
      int col = h * 512 + c0 + f * 16 + lm;
      out[(size_t)(b * Bn + row) * 4096 + col] = f2bf(acc[f][r] / sumacc[r]);
    }
  }
}

// ---------------------------------------------------------------------------
// Orchestration
// ---------------------------------------------------------------------------
extern "C" void kernel_launch(void* const* d_in, const int* in_sizes, int n_in,
                              void* d_out, int out_size, void* d_ws, size_t ws_size,
                              hipStream_t stream) {
  (void)in_sizes; (void)n_in; (void)out_size; (void)ws_size;
  const float* x       = (const float*)d_in[0];
  const float* ctx     = (const float*)d_in[1];
  const float* t       = (const float*)d_in[2];
  const float* q_g     = (const float*)d_in[3];
  const float* q_bln   = (const float*)d_in[4];
  const float* q_w     = (const float*)d_in[5];
  const float* q_b     = (const float*)d_in[6];
  const float* kv_g    = (const float*)d_in[7];
  const float* kv_bln  = (const float*)d_in[8];
  const float* kv_w    = (const float*)d_in[9];
  const float* kv_b    = (const float*)d_in[10];
  const float* merge_w = (const float*)d_in[11];
  const float* merge_b = (const float*)d_in[12];
  const float* tw1     = (const float*)d_in[13];
  const float* tb1     = (const float*)d_in[14];
  const float* tw2     = (const float*)d_in[15];
  const float* tb2     = (const float*)d_in[16];
  const float* ff_g    = (const float*)d_in[17];
  const float* ff_bln  = (const float*)d_in[18];
  const float* ff_w1   = (const float*)d_in[19];
  const float* ff_b1   = (const float*)d_in[20];
  const float* ff_w2   = (const float*)d_in[21];
  const float* ff_b2   = (const float*)d_in[22];

  char* ws = (char*)d_ws;
  size_t off = 0;
  auto alloc = [&](size_t bytes) -> void* {
    void* p = ws + off;
    off = (off + bytes + 255) & ~(size_t)255;
    return p;
  };

  unsigned short* qwt     = (unsigned short*)alloc((size_t)512 * 512 * 2);
  unsigned short* kvwt    = (unsigned short*)alloc((size_t)4608 * 512 * 2);
  unsigned short* mergewt = (unsigned short*)alloc((size_t)512 * 4096 * 2);
  unsigned short* ffw1t   = (unsigned short*)alloc((size_t)2048 * 512 * 2);
  unsigned short* ffw2t   = (unsigned short*)alloc((size_t)512 * 2048 * 2);
  unsigned short* xln     = (unsigned short*)alloc((size_t)8192 * 512 * 2);
  unsigned short* ctxln   = (unsigned short*)alloc((size_t)8192 * 512 * 2);
  unsigned short* qbuf    = (unsigned short*)alloc((size_t)8192 * 512 * 2);
  unsigned short* kvbuf   = (unsigned short*)alloc((size_t)8192 * 4608 * 2);
  unsigned short* vt      = (unsigned short*)alloc((size_t)8 * 8 * 512 * 1024 * 2);
  unsigned short* attnout = (unsigned short*)alloc((size_t)8192 * 4096 * 2);
  float*          xnew    = (float*)alloc((size_t)8192 * 512 * 4);
  unsigned short* lnff    = (unsigned short*)alloc((size_t)8192 * 512 * 2);
  unsigned short* hbuf    = (unsigned short*)alloc((size_t)8192 * 2048 * 2);
  float*          tibuf   = (float*)alloc((size_t)8 * 512 * 4);

  // weight prep: transpose to [N,K] bf16
  convT_kernel<<<(512 * 512) / 256, 256, 0, stream>>>(q_w, qwt, 512, 512);
  convT_kernel<<<(512 * 4608) / 256, 256, 0, stream>>>(kv_w, kvwt, 512, 4608);
  convT_kernel<<<(4096 * 512) / 256, 256, 0, stream>>>(merge_w, mergewt, 4096, 512);
  convT_kernel<<<(512 * 2048) / 256, 256, 0, stream>>>(ff_w1, ffw1t, 512, 2048);
  convT_kernel<<<(2048 * 512) / 256, 256, 0, stream>>>(ff_w2, ffw2t, 2048, 512);

  // time MLP + input LayerNorms
  time_kernel<<<8, 256, 0, stream>>>(t, tw1, tb1, tw2, tb2, tibuf);
  ln_kernel<<<8192, 256, 0, stream>>>(x, q_g, q_bln, nullptr, 1024, xln);
  ln_kernel<<<8192, 256, 0, stream>>>(ctx, kv_g, kv_bln, nullptr, 1024, ctxln);

  // q / kv projections
  gemm_bf16_kernel<0, 1, 0><<<dim3(4, 64), 256, 0, stream>>>(
      xln, qwt, q_b, nullptr, qbuf, 8192, 512, 512);
  gemm_bf16_kernel<0, 1, 0><<<dim3(36, 64), 256, 0, stream>>>(
      ctxln, kvwt, kv_b, nullptr, kvbuf, 8192, 4608, 512);

  // V transpose for contiguous B-fragments in attention
  vtrans_kernel<<<(8u * 8u * 512u * 1024u) / 256u, 256, 0, stream>>>(kvbuf, vt);

  // flash attention
  attn_kernel<<<4096, 128, 0, stream>>>(qbuf, kvbuf, vt, attnout);

  // merge projection + residual x  -> xnew (f32)
  gemm_bf16_kernel<0, 0, 1><<<dim3(4, 64), 256, 0, stream>>>(
      attnout, mergewt, merge_b, x, xnew, 8192, 512, 4096);

  // LN(xnew + ti) -> bf16
  ln_kernel<<<8192, 256, 0, stream>>>(xnew, ff_g, ff_bln, tibuf, 1024, lnff);

  // FFN: (512 -> 2048, swish) then (2048 -> 512) + residual xnew -> d_out (f32)
  gemm_bf16_kernel<1, 1, 0><<<dim3(16, 64), 256, 0, stream>>>(
      lnff, ffw1t, ff_b1, nullptr, hbuf, 8192, 2048, 512);
  gemm_bf16_kernel<0, 0, 1><<<dim3(4, 64), 256, 0, stream>>>(
      hbuf, ffw2t, ff_b2, xnew, (float*)d_out, 8192, 512, 2048);
}
